// FluxPropLayer_50964081934336
// MI455X (gfx1250) — matrix operand order repaired and checked by reading, not compiled, not run on hardware
//
#include <hip/hip_runtime.h>
#include <hip/hip_bf16.h>
#include <math.h>

typedef __bf16 bf16_t;
typedef __attribute__((ext_vector_type(16))) __bf16 v16bf;
typedef __attribute__((ext_vector_type(8)))  __bf16 v8bf;
typedef __attribute__((ext_vector_type(8)))  float  v8f;

#define B_DIM 16384
#define D_DIM 768
#define H_DIM 2048
#define R_DIM 64

// ---------- bf16 helpers (RNE via bit ops; avoids relying on cast support) ----
__device__ __forceinline__ unsigned short f2bfbits(float f) {
  unsigned u = __float_as_uint(f);
  return (unsigned short)((u + 0x7FFFu + ((u >> 16) & 1u)) >> 16);
}
__device__ __forceinline__ bf16_t f2bf(float f) {
  unsigned short s = f2bfbits(f);
  return __builtin_bit_cast(bf16_t, s);
}

// ---------- WMMA fragment loads, contiguous k order k = 16*half + s (wave32) ------------------
// A (16x32, row-major src, lda elems): lane l<16 -> row l, K 0..15;
// lane l>=16 -> row l-16, K 16..31.  aBase pre-offset by row & laneHalf*16.
__device__ __forceinline__ v16bf load_a_frag(const bf16_t* __restrict__ p) {
  v8bf lo = *reinterpret_cast<const v8bf*>(p);        // 16B
  v8bf hi = *reinterpret_cast<const v8bf*>(p + 8);    // 16B, K+8
  return __builtin_shufflevector(lo, hi, 0,1,2,3,4,5,6,7,8,9,10,11,12,13,14,15);
}
// B (32x16): lane l<16 -> col l, K 0..15; lane l>=16 -> col l-16, K 16..31.
// Weight stored row-major [N,K] so col n == contiguous weight row n.
// bBase pre-offset by col & laneHalf*16.
__device__ __forceinline__ v16bf load_b_frag(const bf16_t* __restrict__ p) {
  v8bf lo = *reinterpret_cast<const v8bf*>(p);        // K .. K+7
  v8bf hi = *reinterpret_cast<const v8bf*>(p + 8);    // K+8 .. K+15
  return __builtin_shufflevector(lo, hi, 0,1,2,3,4,5,6,7,8,9,10,11,12,13,14,15);
}
__device__ __forceinline__ v8f wmma_bf16(v16bf a, v16bf b, v8f c) {
  return __builtin_amdgcn_wmma_f32_16x16x32_bf16(false, a, false, b, (short)0, c,
                                                 false, false);
}

// Wave computes 32x64 tile: 2 A-frags x 4 B-frags, 8 WMMAs per 32-wide K step.
__device__ __forceinline__ void gemm_accum(v8f (&acc)[2][4],
    const bf16_t* __restrict__ aBase, size_t lda,
    const bf16_t* __restrict__ bBase, size_t ldb, int K) {
  for (int k = 0; k < K; k += 32) {
    v16bf a0 = load_a_frag(aBase + k);
    v16bf a1 = load_a_frag(aBase + 16 * lda + k);
    v16bf b0 = load_b_frag(bBase + k);
    v16bf b1 = load_b_frag(bBase + 16 * ldb + k);
    v16bf b2 = load_b_frag(bBase + 32 * ldb + k);
    v16bf b3 = load_b_frag(bBase + 48 * ldb + k);
    acc[0][0] = wmma_bf16(a0, b0, acc[0][0]);
    acc[0][1] = wmma_bf16(a0, b1, acc[0][1]);
    acc[0][2] = wmma_bf16(a0, b2, acc[0][2]);
    acc[0][3] = wmma_bf16(a0, b3, acc[0][3]);
    acc[1][0] = wmma_bf16(a1, b0, acc[1][0]);
    acc[1][1] = wmma_bf16(a1, b1, acc[1][1]);
    acc[1][2] = wmma_bf16(a1, b2, acc[1][2]);
    acc[1][3] = wmma_bf16(a1, b3, acc[1][3]);
  }
}

// ---------- fp32 -> bf16 conversion (vectorized x4) ---------------------------
__global__ void cvt_kernel(const float* __restrict__ src,
                           bf16_t* __restrict__ dst, int n4) {
  int idx = blockIdx.x * blockDim.x + threadIdx.x;
  if (idx >= n4) return;
  float4 f = reinterpret_cast<const float4*>(src)[idx];
  ushort4 u;
  u.x = f2bfbits(f.x); u.y = f2bfbits(f.y);
  u.z = f2bfbits(f.z); u.w = f2bfbits(f.w);
  reinterpret_cast<ushort4*>(dst)[idx] = u;
}

// ---------- kernel 1: h_t = (1-sigmoid(alpha))*h_prev + tanh(hW^T + xU^T + b) -
__global__ __launch_bounds__(128) void ht_kernel(
    const bf16_t* __restrict__ hprevbf, const bf16_t* __restrict__ xbf,
    const bf16_t* __restrict__ Wbf, const bf16_t* __restrict__ Uwbf,
    const float* __restrict__ Ub, const float* __restrict__ alpha_raw,
    const float* __restrict__ hprev_f32,
    float* __restrict__ ht_out, bf16_t* __restrict__ htbf) {
  const int lane = threadIdx.x & 31;
  const int wave = threadIdx.x >> 5;
  const int m0 = blockIdx.x * 128 + wave * 32;
  const int n0 = blockIdx.y * 64;
  const int laneN = lane & 15;
  const int laneHalf = lane >> 4;

  const v8f vzero = {0.f, 0.f, 0.f, 0.f, 0.f, 0.f, 0.f, 0.f};
  v8f acc[2][4];
#pragma unroll
  for (int i = 0; i < 2; ++i)
#pragma unroll
    for (int j = 0; j < 4; ++j) acc[i][j] = vzero;

  {  // h_prev @ W^T   (K = H)
    const bf16_t* aBase = hprevbf + (size_t)(m0 + laneN) * H_DIM + laneHalf * 16;
    const bf16_t* bBase = Wbf + (size_t)(n0 + laneN) * H_DIM + laneHalf * 16;
    gemm_accum(acc, aBase, H_DIM, bBase, H_DIM, H_DIM);
  }
  {  // x @ U_w^T      (K = D)
    const bf16_t* aBase = xbf + (size_t)(m0 + laneN) * D_DIM + laneHalf * 16;
    const bf16_t* bBase = Uwbf + (size_t)(n0 + laneN) * D_DIM + laneHalf * 16;
    gemm_accum(acc, aBase, D_DIM, bBase, D_DIM, D_DIM);
  }

#pragma unroll
  for (int ni = 0; ni < 4; ++ni) {
    const int n = n0 + ni * 16 + laneN;
    const float ub = Ub[n];
    const float al = 1.f / (1.f + __expf(-alpha_raw[n]));
#pragma unroll
    for (int mi = 0; mi < 2; ++mi) {
#pragma unroll
      for (int i = 0; i < 8; ++i) {
        const int m = m0 + mi * 16 + laneHalf * 8 + i;
        const size_t o = (size_t)m * H_DIM + n;
        const float liquid = tanhf(acc[mi][ni][i] + ub);
        const float ht = (1.f - al) * hprev_f32[o] + liquid;
        ht_out[o] = ht;
        htbf[o] = f2bf(ht);
      }
    }
  }
}

// ---------- kernel 2: qkv[B,192] = h_t @ [WQ;WK;WV]^T -------------------------
__global__ __launch_bounds__(128) void qkv_kernel(
    const bf16_t* __restrict__ htbf, const bf16_t* __restrict__ wqkvbf,
    float* __restrict__ qkv) {
  const int lane = threadIdx.x & 31;
  const int wave = threadIdx.x >> 5;
  const int m0 = blockIdx.x * 128 + wave * 32;
  const int n0 = blockIdx.y * 64;
  const int laneN = lane & 15;
  const int laneHalf = lane >> 4;

  const v8f vzero = {0.f, 0.f, 0.f, 0.f, 0.f, 0.f, 0.f, 0.f};
  v8f acc[2][4];
#pragma unroll
  for (int i = 0; i < 2; ++i)
#pragma unroll
    for (int j = 0; j < 4; ++j) acc[i][j] = vzero;

  const bf16_t* aBase = htbf + (size_t)(m0 + laneN) * H_DIM + laneHalf * 16;
  const bf16_t* bBase = wqkvbf + (size_t)(n0 + laneN) * H_DIM + laneHalf * 16;
  gemm_accum(acc, aBase, H_DIM, bBase, H_DIM, H_DIM);

#pragma unroll
  for (int ni = 0; ni < 4; ++ni) {
    const int n = n0 + ni * 16 + laneN;
#pragma unroll
    for (int mi = 0; mi < 2; ++mi)
#pragma unroll
      for (int i = 0; i < 8; ++i) {
        const int m = m0 + mi * 16 + laneHalf * 8 + i;
        qkv[(size_t)m * 192 + n] = acc[mi][ni][i];
      }
  }
}

// ---------- kernel 3: per-row scalar attention; gamma folded into routed ------
__global__ __launch_bounds__(256) void attn_kernel(
    const float* __restrict__ qkv, const float* __restrict__ gamma,
    bf16_t* __restrict__ routedbf) {
  __shared__ float sk[4][64];
  __shared__ float sv[4][64];
  const int tid = threadIdx.x;
  const int r = tid >> 6;       // row-within-block 0..3
  const int i = tid & 63;       // output index 0..63
  const int b = blockIdx.x * 4 + r;
  const float* row = qkv + (size_t)b * 192;
  sk[r][i] = row[64 + i];
  sv[r][i] = row[128 + i];
  __syncthreads();
  const float qi = row[i] * 0.125f;  // 1/sqrt(R)
  float mx = -3.0e38f;
  for (int j = 0; j < 64; ++j) mx = fmaxf(mx, qi * sk[r][j]);
  float sum = 0.f, dot = 0.f;
  for (int j = 0; j < 64; ++j) {
    const float e = __expf(qi * sk[r][j] - mx);
    sum += e;
    dot += e * sv[r][j];
  }
  routedbf[(size_t)b * 64 + i] = f2bf(gamma[0] * (dot / sum));
}

// ---------- kernel 4: y = (gamma*routed)@WO^T + x@res_w^T + res_b -------------
__global__ __launch_bounds__(128) void y_kernel(
    const bf16_t* __restrict__ routedbf, const bf16_t* __restrict__ wobf,
    const bf16_t* __restrict__ xbf, const bf16_t* __restrict__ reswbf,
    const float* __restrict__ res_b, float* __restrict__ y_out) {
  const int lane = threadIdx.x & 31;
  const int wave = threadIdx.x >> 5;
  const int m0 = blockIdx.x * 128 + wave * 32;
  const int n0 = blockIdx.y * 64;
  const int laneN = lane & 15;
  const int laneHalf = lane >> 4;

  const v8f vzero = {0.f, 0.f, 0.f, 0.f, 0.f, 0.f, 0.f, 0.f};
  v8f acc[2][4];
#pragma unroll
  for (int i = 0; i < 2; ++i)
#pragma unroll
    for (int j = 0; j < 4; ++j) acc[i][j] = vzero;

  {  // (gamma*routed) @ WO^T  (K = R = 64)
    const bf16_t* aBase = routedbf + (size_t)(m0 + laneN) * R_DIM + laneHalf * 16;
    const bf16_t* bBase = wobf + (size_t)(n0 + laneN) * R_DIM + laneHalf * 16;
    gemm_accum(acc, aBase, R_DIM, bBase, R_DIM, R_DIM);
  }
  {  // x @ res_w^T            (K = D)
    const bf16_t* aBase = xbf + (size_t)(m0 + laneN) * D_DIM + laneHalf * 16;
    const bf16_t* bBase = reswbf + (size_t)(n0 + laneN) * D_DIM + laneHalf * 16;
    gemm_accum(acc, aBase, D_DIM, bBase, D_DIM, D_DIM);
  }

#pragma unroll
  for (int ni = 0; ni < 4; ++ni) {
    const int n = n0 + ni * 16 + laneN;
    const float rb = res_b[n];
#pragma unroll
    for (int mi = 0; mi < 2; ++mi)
#pragma unroll
      for (int i = 0; i < 8; ++i) {
        const int m = m0 + mi * 16 + laneHalf * 8 + i;
        y_out[(size_t)m * H_DIM + n] = acc[mi][ni][i] + rb;
      }
  }
}

// ---------- launcher ----------------------------------------------------------
extern "C" void kernel_launch(void* const* d_in, const int* in_sizes, int n_in,
                              void* d_out, int out_size, void* d_ws,
                              size_t ws_size, hipStream_t stream) {
  const float* h_prev    = (const float*)d_in[0];
  const float* x_t       = (const float*)d_in[1];
  const float* W         = (const float*)d_in[2];
  const float* U_w       = (const float*)d_in[3];
  const float* U_b       = (const float*)d_in[4];
  const float* alpha_raw = (const float*)d_in[5];
  const float* WQ        = (const float*)d_in[6];
  const float* WK        = (const float*)d_in[7];
  const float* WV        = (const float*)d_in[8];
  const float* WO        = (const float*)d_in[9];
  const float* res_w     = (const float*)d_in[10];
  const float* res_b     = (const float*)d_in[11];
  const float* gamma     = (const float*)d_in[12];

  float* ht_out = (float*)d_out;
  float* y_out  = ht_out + (size_t)B_DIM * H_DIM;

  char* ws = (char*)d_ws;
  size_t off = 0;
  auto carve = [&](size_t bytes) -> char* {
    char* p = ws + off;
    off += (bytes + 255) & ~(size_t)255;
    return p;
  };
  bf16_t* hprevbf  = (bf16_t*)carve((size_t)B_DIM * H_DIM * 2);
  bf16_t* xbf      = (bf16_t*)carve((size_t)B_DIM * D_DIM * 2);
  bf16_t* htbf     = (bf16_t*)carve((size_t)B_DIM * H_DIM * 2);
  bf16_t* routedbf = (bf16_t*)carve((size_t)B_DIM * R_DIM * 2);
  bf16_t* Wbf      = (bf16_t*)carve((size_t)H_DIM * H_DIM * 2);
  bf16_t* Uwbf     = (bf16_t*)carve((size_t)H_DIM * D_DIM * 2);
  bf16_t* reswbf   = (bf16_t*)carve((size_t)H_DIM * D_DIM * 2);
  bf16_t* wqkvbf   = (bf16_t*)carve((size_t)3 * R_DIM * H_DIM * 2);
  bf16_t* wobf     = (bf16_t*)carve((size_t)H_DIM * R_DIM * 2);
  float*  qkv      = (float*)carve((size_t)B_DIM * 192 * 4);

  auto cvt = [&](const float* src, bf16_t* dst, size_t n) {
    int n4 = (int)(n / 4);
    cvt_kernel<<<(n4 + 255) / 256, 256, 0, stream>>>(src, dst, n4);
  };
  cvt(h_prev, hprevbf, (size_t)B_DIM * H_DIM);
  cvt(x_t, xbf, (size_t)B_DIM * D_DIM);
  cvt(W, Wbf, (size_t)H_DIM * H_DIM);
  cvt(U_w, Uwbf, (size_t)H_DIM * D_DIM);
  cvt(res_w, reswbf, (size_t)H_DIM * D_DIM);
  cvt(WQ, wqkvbf, (size_t)R_DIM * H_DIM);
  cvt(WK, wqkvbf + (size_t)R_DIM * H_DIM, (size_t)R_DIM * H_DIM);
  cvt(WV, wqkvbf + (size_t)2 * R_DIM * H_DIM, (size_t)R_DIM * H_DIM);
  cvt(WO, wobf, (size_t)H_DIM * R_DIM);

  const dim3 blk(128);
  ht_kernel<<<dim3(B_DIM / 128, H_DIM / 64), blk, 0, stream>>>(
      hprevbf, xbf, Wbf, Uwbf, U_b, alpha_raw, h_prev, ht_out, htbf);
  qkv_kernel<<<dim3(B_DIM / 128, 3), blk, 0, stream>>>(htbf, wqkvbf, qkv);
  attn_kernel<<<B_DIM / 4, 256, 0, stream>>>(qkv, gamma, routedbf);
  y_kernel<<<dim3(B_DIM / 128, H_DIM / 64), blk, 0, stream>>>(
      routedbf, wobf, xbf, reswbf, res_b, y_out);
}
